// GeometricSDEContiformer_78898549227684
// MI455X (gfx1250) — matrix operand order repaired and checked by reading, not compiled, not run on hardware
//
#include <hip/hip_runtime.h>

typedef __bf16 bf16;
typedef __attribute__((ext_vector_type(16))) __bf16 v16bf;
typedef __attribute__((ext_vector_type(8)))  float   v8f;
typedef int v4i_vs __attribute__((vector_size(16)));   // matches builtin's pointee type

#define EPSF 1e-8f
#define Bb  64
#define Ss  512
#define INd 3
#define Hh  128
#define Dd  512
#define HDd 64
#define Cc  5
#define TOK (Bb*Ss)   // 32768

// ======================= CDNA5 async global->LDS copy (ASYNCcnt path) =======================

#if defined(__gfx1250__) && __has_builtin(__builtin_amdgcn_global_load_async_to_lds_b128) && \
    __has_builtin(__builtin_amdgcn_s_wait_asynccnt)
#define USE_ASYNC_LDS 1
#else
#define USE_ASYNC_LDS 0
#endif

__device__ inline void cp_async16(void* lds_dst, const void* gsrc) {
#if USE_ASYNC_LDS
  __builtin_amdgcn_global_load_async_to_lds_b128(
      (__attribute__((address_space(1))) v4i_vs*)(gsrc),
      (__attribute__((address_space(3))) v4i_vs*)(lds_dst), 0, 0);
#else
  *(uint4*)lds_dst = *(const uint4*)gsrc;
#endif
}
__device__ inline void async_wait_2() {
#if USE_ASYNC_LDS
  __builtin_amdgcn_s_wait_asynccnt(2);
#endif
}
__device__ inline void async_wait_0() {
#if USE_ASYNC_LDS
  __builtin_amdgcn_s_wait_asynccnt(0);
#endif
}

__device__ inline float fast_tanh(float x) {
#if __has_builtin(__builtin_amdgcn_tanhf)
  return __builtin_amdgcn_tanhf(x);
#elif __has_builtin(__builtin_amdgcn_tanh_f32)
  return __builtin_amdgcn_tanh_f32(x);
#else
  return tanhf(x);
#endif
}

// ======================= WMMA fragment helpers (ISA 7.12.2) =======================

__device__ inline v8f zero8() {
  v8f z;
#pragma unroll
  for (int i = 0; i < 8; ++i) z[i] = 0.f;
  return z;
}

__device__ inline v8f wmma_bf16(v16bf a, v16bf b, v8f c) {
  return __builtin_amdgcn_wmma_f32_16x16x32_bf16(false, a, false, b, (short)0, c, false, false);
}

// A fragment from row-major bf16 matrix: element (m,k); lane = (m&15) + 16*(k>>4)
__device__ inline v16bf ld_a_bf(const bf16* A, int lda, int m0, int k0) {
  int lane = threadIdx.x & 31;
  const bf16* p = A + (long)(m0 + (lane & 15)) * lda + k0 + ((lane >> 4) << 4);
  v16bf a;
#pragma unroll
  for (int e = 0; e < 16; ++e) a[e] = p[e];
  return a;
}

// same, fp32 source (global or LDS) converted to bf16 in-register
__device__ inline v16bf ld_a_f32(const float* A, int lda, int m0, int k0) {
  int lane = threadIdx.x & 31;
  const float* p = A + (long)(m0 + (lane & 15)) * lda + k0 + ((lane >> 4) << 4);
  v16bf a;
#pragma unroll
  for (int e = 0; e < 16; ++e) a[e] = (bf16)p[e];
  return a;
}

// B fragment from TRANSPOSED layout Bt[n][k] (N x K row-major): lane = (n&15)+16*(k>>4)
__device__ inline v16bf ld_b_t(const bf16* Bt, int ldb, int n0, int k0) {
  int lane = threadIdx.x & 31;
  const bf16* p = Bt + (long)(n0 + (lane & 15)) * ldb + k0 + ((lane >> 4) << 4);
  v16bf b;
#pragma unroll
  for (int e = 0; e < 16; ++e) b[e] = p[e];
  return b;
}

// ======================= weight transpose+cvt: wt[n*Kpad+k] = w[k*N+n] =======================

__global__ void transpose_cvt(const float* __restrict__ w, bf16* __restrict__ wt,
                              int K, int N, int Kpad) {
  int n = blockIdx.x;
  int k = blockIdx.y * 256 + threadIdx.x;
  if (k < Kpad) wt[(long)n * Kpad + k] = (k < K) ? (bf16)w[(long)k * N + n] : (bf16)0.f;
}

// ======================= feature encoder -> y0 (only s==0 of enc is used) =======================

__global__ void feat_y0(const float* __restrict__ ts, const float* __restrict__ few,
                        const float* __restrict__ feb, const float* __restrict__ lng,
                        const float* __restrict__ lnb, float* __restrict__ sde) {
  __shared__ float red[128];
  int b = blockIdx.x, h = threadIdx.x;
  const float* tsr = ts + (long)b * Ss * INd;   // s = 0
  float e = feb[h] + tsr[0] * few[0 * Hh + h] + tsr[1] * few[1 * Hh + h] + tsr[2] * few[2 * Hh + h];
  red[h] = e; __syncthreads();
  for (int o = 64; o > 0; o >>= 1) { if (h < o) red[h] += red[h + o]; __syncthreads(); }
  float mean = red[0] / 128.f; __syncthreads();
  float d = e - mean;
  red[h] = d * d; __syncthreads();
  for (int o = 64; o > 0; o >>= 1) { if (h < o) red[h] += red[h + o]; __syncthreads(); }
  float var = red[0] / 128.f;
  float v = d * rsqrtf(var + 1e-5f) * lng[h] + lnb[h];
  float r = fmaxf(v, 0.f);
  float sp = (r > 20.f) ? r : log1pf(expf(r));
  sde[((long)b * Ss) * Hh + h] = sp + 2.f * EPSF;
}

// ======================= SDE scan: persistent single-workgroup, LDS-resident state ==============

__device__ void mm64(const bf16* Als, int lda, const bf16* __restrict__ Bt, int ldb,
                     const float* __restrict__ bias, int N, int K,
                     int tanh_mode, bf16* outb, float* outf) {
  int wave = threadIdx.x >> 5, lane = threadIdx.x & 31;
  int Nt = N >> 4;
  int tpw = (Nt * 4) >> 3;   // tiles per wave
  int nl = lane & 15, mh = (lane >> 4) * 8;
  for (int i = 0; i < tpw; ++i) {
    int t = wave * tpw + i;
    int mt = t & 3, nt = t >> 2;
    v8f acc = zero8();
    for (int k0 = 0; k0 < K; k0 += 32) {
      v16bf a = ld_a_bf(Als, lda, mt * 16, k0);
      v16bf b = ld_b_t(Bt, ldb, nt * 16, k0);
      acc = wmma_bf16(a, b, acc);
    }
    int n = nt * 16 + nl;
    float bv = bias[n];
#pragma unroll
    for (int r = 0; r < 8; ++r) {
      int m = mt * 16 + mh + r;
      float v = acc[r] + bv;
      if (tanh_mode) outb[m * N + n] = (bf16)fast_tanh(v);
      else           outf[m * N + n] = v;
    }
  }
}

__global__ void sde_scan(const float* __restrict__ times, const float* __restrict__ noise,
                         const bf16* mw1, const float* mb1, const bf16* mw2, const float* mb2,
                         const bf16* mw3, const float* mb3,
                         const bf16* sw1, const float* sb1, const bf16* sw2, const float* sb2,
                         const bf16* sw3, const float* sb3,
                         const float* pmin, const float* pmax, float* __restrict__ sde) {
  extern __shared__ char smem_raw[];
  float* yf = (float*)smem_raw;        // 64*128
  float* mu = yf + 8192;               // 64*128
  float* sg = mu + 8192;               // 64*128
  bf16* ty = (bf16*)(sg + 8192);       // 64*160 (K padded 129->160)
  bf16* h1 = ty + 64 * 160;            // 64*256
  bf16* h2 = h1 + 64 * 256;            // 64*128
  int tid = threadIdx.x;
  float smin = fabsf(pmin[0]), smax = fabsf(pmax[0]);

  for (int i = tid; i < 8192; i += 256) {
    int bq = i >> 7, k = i & 127;
    yf[i] = sde[((long)bq * Ss) * Hh + k];
  }
  for (int i = tid; i < 64 * 32; i += 256) {
    int m = i >> 5, c = 128 + (i & 31);
    ty[m * 160 + c] = (bf16)0.f;
  }
  __syncthreads();

  for (int s = 1; s < Ss; ++s) {
    float tp = times[s - 1], tc = times[s];
    float dt = tc - tp, sq = sqrtf(dt);
    for (int i = tid; i < 8192; i += 256) {
      int m = i >> 7, k = i & 127;
      ty[m * 160 + k] = (bf16)(fabsf(yf[i]) + EPSF);
    }
    if (tid < 64) ty[tid * 160 + 128] = (bf16)tp;
    __syncthreads();
    mm64(ty, 160, mw1, 160, mb1, 256, 160, 1, h1, nullptr); __syncthreads();
    mm64(h1, 256, mw2, 256, mb2, 128, 256, 1, h2, nullptr); __syncthreads();
    mm64(h2, 128, mw3, 128, mb3, 128, 128, 0, nullptr, mu); __syncthreads();
    mm64(ty, 160, sw1, 160, sb1, 256, 160, 1, h1, nullptr); __syncthreads();
    mm64(h1, 256, sw2, 256, sb2, 128, 256, 1, h2, nullptr); __syncthreads();
    mm64(h2, 128, sw3, 128, sb3, 128, 128, 0, nullptr, sg); __syncthreads();
    for (int i = tid; i < 8192; i += 256) {
      int bq = i >> 7, k = i & 127;
      float y  = yf[i];
      float ys = fabsf(y) + EPSF;
      float m_ = mu[i];
      float s_ = fminf(fmaxf(sg[i], smin), smax);
      float dw = noise[((long)bq * Ss + s) * Hh + k];
      float yn = y + m_ * ys * dt + s_ * ys * sq * dw;
      yn = fabsf(yn) + EPSF;
      yf[i] = yn;
      sde[((long)bq * Ss + s) * Hh + k] = yn;
    }
    __syncthreads();
  }
}

// ======================= generic WMMA GEMM with async-LDS B staging ===========================
// 128x128 block tile, 256 threads; B chunk (128x32 bf16, 8KB) double-buffered in LDS via
// GLOBAL_LOAD_ASYNC_TO_LDS_B128; wave w owns rows [w*16, w*16+16) x 128 cols.
// cbt!=0: write Cb in transposed attention-V layout vt[(b*8+h)][hd][s] (contiguous stores).

__global__ void gemm_wmma(const float* __restrict__ Af, const bf16* __restrict__ Ab, int lda,
                          const bf16* __restrict__ Bt, int ldb,
                          const float* __restrict__ bias,
                          float* __restrict__ Cf, bf16* __restrict__ Cb, int ldc,
                          int M, int N, int K, int relu, int cbt) {
  __shared__ bf16 Bs[2][128 * 32];   // 16 KB
  int tid = threadIdx.x;
  int wave = tid >> 5, lane = tid & 31;
  int m0 = blockIdx.y * 128 + wave * 16;
  int n0 = blockIdx.x * 128;
  int crow = tid >> 1, ccol = (tid & 1) * 16;   // each thread copies 16 bf16 (2 x 16B)

  v8f acc[8];
#pragma unroll
  for (int t = 0; t < 8; ++t) acc[t] = zero8();

  { // prime buffer 0 with k-chunk 0
    const bf16* s0 = Bt + (long)(n0 + crow) * ldb + ccol;
    bf16* d0 = &Bs[0][0] + crow * 32 + ccol;
    cp_async16(d0, s0); cp_async16(d0 + 8, s0 + 8);
  }
  int buf = 0;
  for (int k0 = 0; k0 < K; k0 += 32) {
    if (k0 + 32 < K) {
      const bf16* s0 = Bt + (long)(n0 + crow) * ldb + (k0 + 32) + ccol;
      bf16* d0 = &Bs[buf ^ 1][0] + crow * 32 + ccol;
      cp_async16(d0, s0); cp_async16(d0 + 8, s0 + 8);
      async_wait_2();                 // current chunk's 2 per-wave async ops done
    } else {
      async_wait_0();
    }
    __syncthreads();
    v16bf a = Ab ? ld_a_bf(Ab, lda, m0, k0) : ld_a_f32(Af, lda, m0, k0);
#pragma unroll
    for (int t = 0; t < 8; ++t) {
      v16bf b = ld_b_t(&Bs[buf][0], 32, t * 16, 0);
      acc[t] = wmma_bf16(a, b, acc[t]);
    }
    __syncthreads();                  // all waves done reading buf before it is rewritten
    buf ^= 1;
  }

  int nl = lane & 15, mh = (lane >> 4) * 8;
#pragma unroll
  for (int t = 0; t < 8; ++t) {
    int n = n0 + t * 16 + nl;
    float bv = bias ? bias[n] : 0.f;
#pragma unroll
    for (int r = 0; r < 8; ++r) {
      float v = acc[t][r] + bv;
      if (relu) v = fmaxf(v, 0.f);
      int m = m0 + mh + r;
      if (cbt) {
        // token m = (b,s), col n = (h,hd) -> vt[((b*8+h)*64+hd)*512 + s]
        long off = ((long)((m >> 9) * 8 + (n >> 6)) * 64 + (n & 63)) * 512 + (m & 511);
        Cb[off] = (bf16)v;
      } else {
        long off = (long)m * ldc + n;
        if (Cf) Cf[off] = v;
        if (Cb) Cb[off] = (bf16)v;
      }
    }
  }
}

// ======================= attention: one WG per (batch, head, 64-query chunk) ====================

__global__ void attn_wmma(const bf16* __restrict__ Q, const bf16* __restrict__ Kc,
                          const bf16* __restrict__ Vt, bf16* __restrict__ O) {
  extern __shared__ char smem_raw[];
  float* sS   = (float*)smem_raw;   // 64 x 512 scores
  float* red  = sS + 64 * Ss;       // 256
  float* rowv = red + 256;          // 64
  int qc = blockIdx.x, h = blockIdx.y, b = blockIdx.z;
  int wave = threadIdx.x >> 5, lane = threadIdx.x & 31;
  const bf16* Qb  = Q  + ((long)(b * Ss + qc * 64)) * Dd + h * HDd;
  const bf16* Kb  = Kc + ((long)b * Ss) * Dd + h * HDd;
  const bf16* VbT = Vt + ((long)(b * 8 + h)) * HDd * Ss;

  { // phase 1: S = (Q K^T) * scale   (4 m-tiles x 32 n-tiles, 16 tiles/wave)
    int mt = wave & 3;
    int ntbase = (wave >> 2) * 16;
    v16bf a0 = ld_a_bf(Qb, Dd, mt * 16, 0);
    v16bf a1 = ld_a_bf(Qb, Dd, mt * 16, 32);
    int nl = lane & 15, mh = (lane >> 4) * 8;
    for (int j = 0; j < 16; ++j) {
      int nt = ntbase + j;
      v8f acc = zero8();
      acc = wmma_bf16(a0, ld_b_t(Kb, Dd, nt * 16, 0),  acc);
      acc = wmma_bf16(a1, ld_b_t(Kb, Dd, nt * 16, 32), acc);
      int n = nt * 16 + nl;
#pragma unroll
      for (int r = 0; r < 8; ++r)
        sS[(mt * 16 + mh + r) * Ss + n] = acc[r] * 0.125f;  // 1/sqrt(64)
    }
  }
  __syncthreads();

  { // phase 2: row softmax, 4 threads per row
    int row = threadIdx.x >> 2, part = threadIdx.x & 3;
    float* pr = sS + row * Ss + part * 128;
    float mx = -3.4e38f;
    for (int i = 0; i < 128; ++i) mx = fmaxf(mx, pr[i]);
    red[threadIdx.x] = mx; __syncthreads();
    if (part == 0)
      rowv[row] = fmaxf(fmaxf(red[row * 4], red[row * 4 + 1]),
                        fmaxf(red[row * 4 + 2], red[row * 4 + 3]));
    __syncthreads();
    float rm = rowv[row];
    float s = 0.f;
    for (int i = 0; i < 128; ++i) { float e = __expf(pr[i] - rm); pr[i] = e; s += e; }
    red[threadIdx.x] = s; __syncthreads();
    if (part == 0)
      rowv[row] = red[row * 4] + red[row * 4 + 1] + red[row * 4 + 2] + red[row * 4 + 3];
    __syncthreads();
    float inv = 1.f / rowv[row];
    for (int i = 0; i < 128; ++i) pr[i] *= inv;
  }
  __syncthreads();

  { // phase 3: O = P @ V  (4 m-tiles x 4 n-tiles, 2 tiles/wave)
    int mt = wave & 3;
    int ntb = (wave >> 2) * 2;
    v8f acc[2];
    acc[0] = zero8(); acc[1] = zero8();
    for (int k0 = 0; k0 < Ss; k0 += 32) {
      v16bf a = ld_a_f32(sS, Ss, mt * 16, k0);
#pragma unroll
      for (int j = 0; j < 2; ++j)
        acc[j] = wmma_bf16(a, ld_b_t(VbT, Ss, (ntb + j) * 16, k0), acc[j]);
    }
    int nl = lane & 15, mh = (lane >> 4) * 8;
#pragma unroll
    for (int j = 0; j < 2; ++j)
#pragma unroll
      for (int r = 0; r < 8; ++r) {
        int srow = qc * 64 + mt * 16 + mh + r;
        O[((long)(b * Ss + srow)) * Dd + h * HDd + (ntb + j) * 16 + nl] = (bf16)acc[j][r];
      }
  }
}

// ======================= residual + LayerNorm (fp32 master + bf16 shadow) =======================

__global__ void add_ln(float* __restrict__ x, const float* __restrict__ t,
                       const float* __restrict__ g, const float* __restrict__ be,
                       bf16* __restrict__ xb) {
  __shared__ float red[256];
  long row = blockIdx.x;
  float* xr = x + row * Dd;
  const float* tr = t + row * Dd;
  int tid = threadIdx.x;
  float v0 = xr[tid] + tr[tid];
  float v1 = xr[tid + 256] + tr[tid + 256];
  red[tid] = v0 + v1; __syncthreads();
  for (int o = 128; o > 0; o >>= 1) { if (tid < o) red[tid] += red[tid + o]; __syncthreads(); }
  float mean = red[0] / 512.f; __syncthreads();
  red[tid] = (v0 - mean) * (v0 - mean) + (v1 - mean) * (v1 - mean); __syncthreads();
  for (int o = 128; o > 0; o >>= 1) { if (tid < o) red[tid] += red[tid + o]; __syncthreads(); }
  float inv = rsqrtf(red[0] / 512.f + 1e-5f);
  float o0 = (v0 - mean) * inv * g[tid] + be[tid];
  float o1 = (v1 - mean) * inv * g[tid + 256] + be[tid + 256];
  xr[tid] = o0; xr[tid + 256] = o1;
  xb[row * Dd + tid] = (bf16)o0; xb[row * Dd + tid + 256] = (bf16)o1;
}

// ======================= pooling + classifier (tiny, scalar) =======================

__global__ void pool_mean(const float* __restrict__ x, float* __restrict__ pooled) {
  int b = blockIdx.x, tid = threadIdx.x;
  for (int d = tid; d < Dd; d += 256) {
    float s = 0.f;
    for (int ss = 0; ss < Ss; ++ss) s += x[((long)b * Ss + ss) * Dd + d];
    pooled[b * Dd + d] = s / 512.f;
  }
}

__global__ void classifier(const float* __restrict__ pooled,
                           const float* __restrict__ w1, const float* __restrict__ b1,
                           const float* __restrict__ w2, const float* __restrict__ b2,
                           float* __restrict__ logits) {
  __shared__ float hsh[256];
  int b = blockIdx.x, j = threadIdx.x;
  const float* pr = pooled + b * Dd;
  float acc = b1[j];
  for (int d = 0; d < Dd; ++d) acc += pr[d] * w1[d * 256 + j];
  hsh[j] = fmaxf(acc, 0.f);
  __syncthreads();
  if (j < Cc) {
    float lg = b2[j];
    for (int q = 0; q < 256; ++q) lg += hsh[q] * w2[q * Cc + j];
    logits[b * Cc + j] = lg;
  }
}

// ======================= host driver =======================

extern "C" void kernel_launch(void* const* d_in, const int* in_sizes, int n_in,
                              void* d_out, int out_size, void* d_ws, size_t ws_size,
                              hipStream_t stream) {
  (void)in_sizes; (void)n_in; (void)out_size; (void)ws_size;
  int idx = 0;
  const float* ts     = (const float*)d_in[idx++];
  const float* times  = (const float*)d_in[idx++];
  const float* noise  = (const float*)d_in[idx++];
  const float* fe_w   = (const float*)d_in[idx++];
  const float* fe_b   = (const float*)d_in[idx++];
  const float* fe_lng = (const float*)d_in[idx++];
  const float* fe_lnb = (const float*)d_in[idx++];
  struct M3 { const float *w1, *b1, *w2, *b2, *w3, *b3; } mu, sg;
  mu.w1 = (const float*)d_in[idx++]; mu.b1 = (const float*)d_in[idx++];
  mu.w2 = (const float*)d_in[idx++]; mu.b2 = (const float*)d_in[idx++];
  mu.w3 = (const float*)d_in[idx++]; mu.b3 = (const float*)d_in[idx++];
  sg.w1 = (const float*)d_in[idx++]; sg.b1 = (const float*)d_in[idx++];
  sg.w2 = (const float*)d_in[idx++]; sg.b2 = (const float*)d_in[idx++];
  sg.w3 = (const float*)d_in[idx++]; sg.b3 = (const float*)d_in[idx++];
  const float* min_sig = (const float*)d_in[idx++];
  const float* max_sig = (const float*)d_in[idx++];
  const float* cf_in_w = (const float*)d_in[idx++];
  const float* cf_in_b = (const float*)d_in[idx++];
  struct Layer { const float *wq,*bq,*wk,*bk,*wv,*bv,*wo,*bo,*g1,*be1,*g2,*be2,*fw1,*fb1,*fw2,*fb2; } L[4];
  for (int l = 0; l < 4; ++l) {
    L[l].wq  = (const float*)d_in[idx++]; L[l].bq  = (const float*)d_in[idx++];
    L[l].wk  = (const float*)d_in[idx++]; L[l].bk  = (const float*)d_in[idx++];
    L[l].wv  = (const float*)d_in[idx++]; L[l].bv  = (const float*)d_in[idx++];
    L[l].wo  = (const float*)d_in[idx++]; L[l].bo  = (const float*)d_in[idx++];
    L[l].g1  = (const float*)d_in[idx++]; L[l].be1 = (const float*)d_in[idx++];
    L[l].g2  = (const float*)d_in[idx++]; L[l].be2 = (const float*)d_in[idx++];
    L[l].fw1 = (const float*)d_in[idx++]; L[l].fb1 = (const float*)d_in[idx++];
    L[l].fw2 = (const float*)d_in[idx++]; L[l].fb2 = (const float*)d_in[idx++];
  }
  const float* cls_w1 = (const float*)d_in[idx++];
  const float* cls_b1 = (const float*)d_in[idx++];
  const float* cls_w2 = (const float*)d_in[idx++];
  const float* cls_b2 = (const float*)d_in[idx++];

  float* out = (float*)d_out;
  float* logits = out;            // 64*5
  float* sde = out + Bb * Cc;     // 64*512*128

  char* cur = (char*)d_ws;
  auto alloc = [&](size_t bytes) -> void* {
    void* p = (void*)cur;
    cur += (bytes + 255) & ~(size_t)255;
    return p;
  };
  bf16* mw1t = (bf16*)alloc(256 * 160 * 2);
  bf16* mw2t = (bf16*)alloc(128 * 256 * 2);
  bf16* mw3t = (bf16*)alloc(128 * 128 * 2);
  bf16* sw1t = (bf16*)alloc(256 * 160 * 2);
  bf16* sw2t = (bf16*)alloc(128 * 256 * 2);
  bf16* sw3t = (bf16*)alloc(128 * 128 * 2);
  bf16* cfwt = (bf16*)alloc((size_t)512 * 128 * 2);
  bf16 *wqt[4], *wkt[4], *wvt[4], *wot[4], *f1t[4], *f2t[4];
  for (int l = 0; l < 4; ++l) {
    wqt[l] = (bf16*)alloc((size_t)512 * 512 * 2);
    wkt[l] = (bf16*)alloc((size_t)512 * 512 * 2);
    wvt[l] = (bf16*)alloc((size_t)512 * 512 * 2);
    wot[l] = (bf16*)alloc((size_t)512 * 512 * 2);
    f1t[l] = (bf16*)alloc((size_t)2048 * 512 * 2);
    f2t[l] = (bf16*)alloc((size_t)512 * 2048 * 2);
  }
  float* xf = (float*)alloc((size_t)TOK * Dd * 4);
  bf16*  xb = (bf16*) alloc((size_t)TOK * Dd * 2);
  float* tf = (float*)alloc((size_t)TOK * Dd * 4);
  char* uni = (char*)alloc((size_t)TOK * 2048 * 2);   // {q,k,v,o} <-> FFN hidden
  bf16* qb = (bf16*)uni;
  bf16* kb = qb + (size_t)TOK * Dd;
  bf16* vt = kb + (size_t)TOK * Dd;   // V in transposed (b,h,hd,s) layout
  bf16* ob = vt + (size_t)TOK * Dd;
  bf16* hb = (bf16*)uni;
  float* pooled = (float*)alloc((size_t)Bb * Dd * 4);

  auto tr = [&](const float* src, bf16* dst, int K, int N, int Kpad) {
    dim3 g(N, (Kpad + 255) / 256);
    transpose_cvt<<<g, 256, 0, stream>>>(src, dst, K, N, Kpad);
  };
  tr(mu.w1, mw1t, 129, 256, 160); tr(mu.w2, mw2t, 256, 128, 256); tr(mu.w3, mw3t, 128, 128, 128);
  tr(sg.w1, sw1t, 129, 256, 160); tr(sg.w2, sw2t, 256, 128, 256); tr(sg.w3, sw3t, 128, 128, 128);
  tr(cf_in_w, cfwt, 128, 512, 128);
  for (int l = 0; l < 4; ++l) {
    tr(L[l].wq, wqt[l], 512, 512, 512);  tr(L[l].wk, wkt[l], 512, 512, 512);
    tr(L[l].wv, wvt[l], 512, 512, 512);  tr(L[l].wo, wot[l], 512, 512, 512);
    tr(L[l].fw1, f1t[l], 512, 2048, 512); tr(L[l].fw2, f2t[l], 2048, 512, 2048);
  }

  feat_y0<<<Bb, 128, 0, stream>>>(ts, fe_w, fe_b, fe_lng, fe_lnb, sde);
  size_t scan_lds = (size_t)(8192 * 3) * 4 + (size_t)(64 * 160 + 64 * 256 + 64 * 128) * 2;
  sde_scan<<<1, 256, scan_lds, stream>>>(times, noise,
      mw1t, mu.b1, mw2t, mu.b2, mw3t, mu.b3,
      sw1t, sg.b1, sw2t, sg.b2, sw3t, sg.b3,
      min_sig, max_sig, sde);

  gemm_wmma<<<dim3(Dd / 128, TOK / 128), 256, 0, stream>>>(
      sde, nullptr, Hh, cfwt, Hh, cf_in_b, xf, xb, Dd, TOK, Dd, Hh, 0, 0);

  size_t attn_lds = (size_t)(64 * Ss + 256 + 64) * 4;
  for (int l = 0; l < 4; ++l) {
    gemm_wmma<<<dim3(4, TOK / 128), 256, 0, stream>>>(nullptr, xb, Dd, wqt[l], Dd, L[l].bq,
                                                      nullptr, qb, Dd, TOK, Dd, Dd, 0, 0);
    gemm_wmma<<<dim3(4, TOK / 128), 256, 0, stream>>>(nullptr, xb, Dd, wkt[l], Dd, L[l].bk,
                                                      nullptr, kb, Dd, TOK, Dd, Dd, 0, 0);
    gemm_wmma<<<dim3(4, TOK / 128), 256, 0, stream>>>(nullptr, xb, Dd, wvt[l], Dd, L[l].bv,
                                                      nullptr, vt, Dd, TOK, Dd, Dd, 0, 1);
    attn_wmma<<<dim3(8, 8, Bb), 256, attn_lds, stream>>>(qb, kb, vt, ob);
    gemm_wmma<<<dim3(4, TOK / 128), 256, 0, stream>>>(nullptr, ob, Dd, wot[l], Dd, L[l].bo,
                                                      tf, nullptr, Dd, TOK, Dd, Dd, 0, 0);
    add_ln<<<TOK, 256, 0, stream>>>(xf, tf, L[l].g1, L[l].be1, xb);
    gemm_wmma<<<dim3(16, TOK / 128), 256, 0, stream>>>(nullptr, xb, Dd, f1t[l], Dd, L[l].fb1,
                                                       nullptr, hb, 2048, TOK, 2048, Dd, 1, 0);
    gemm_wmma<<<dim3(4, TOK / 128), 256, 0, stream>>>(nullptr, hb, 2048, f2t[l], 2048, L[l].fb2,
                                                      tf, nullptr, Dd, TOK, Dd, 2048, 0, 0);
    add_ln<<<TOK, 256, 0, stream>>>(xf, tf, L[l].g2, L[l].be2, xb);
  }

  pool_mean<<<Bb, 256, 0, stream>>>(xf, pooled);
  classifier<<<Bb, 256, 0, stream>>>(pooled, cls_w1, cls_b1, cls_w2, cls_b2, logits);
}
